// PointConvNet_32658931319631
// MI455X (gfx1250) — compile-verified, matrix operand order checked by reading, not run on hardware
//
#include <hip/hip_runtime.h>

// ---------------------------------------------------------------------------
// PointConv (PyG-style) for MI455X / gfx1250:
//   msg = relu(concat(x[src], pos[src]-pos[dst]) @ W1 + b1) @ W2 + b2
//   out[i] = segment_max over dst==i (self-loops included)
// bf16 WMMA (16x16x32) with f32 accumulate; weights packed to fragment layout.
// ---------------------------------------------------------------------------

typedef __attribute__((ext_vector_type(16))) __bf16 v16bf;
typedef __attribute__((ext_vector_type(8)))  __bf16 v8bf;
typedef __attribute__((ext_vector_type(8)))  float  v8f;

#define N_NODES 50000
#define E_EDGES 800000
#define M_TOTAL (E_EDGES + N_NODES)   // 850000 messages (edges + self loops)
#define NTILES  (M_TOTAL / 16)        // 53125 (exact)
#define DIN     64
#define HID     64

#define FRAG_ELEMS 512                // 32 lanes x 16 bf16 per B fragment
#define W1FRAGS 12                    // 4 N-tiles x 3 K-chunks (K padded 67->96)
#define W2FRAGS 8                     // 4 N-tiles x 2 K-chunks (K=64)
#define W1_ELEMS (W1FRAGS * FRAG_ELEMS)   // 6144
#define W2_ELEMS (W2FRAGS * FRAG_ELEMS)   // 4096
#define WTOTAL   (W1_ELEMS + W2_ELEMS)    // 10240 bf16 = 20KB

// ---------------------------------------------------------------------------
// Pack W1/W2 (row-major f32 [K][N]) into WMMA 16-bit B-fragment layout (bf16).
// 16-bit operand striping (ISA 7.12.2): lane<16 holds K {b..b+7, b+16..b+23}
// with b = 32c; lane>=16 holds b = 32c+8.  Element e<8 -> K=b+e, e>=8 -> K=b+8+e.
// ---------------------------------------------------------------------------
__global__ void pack_weights_kernel(const float* __restrict__ W1,
                                    const float* __restrict__ W2,
                                    __bf16* __restrict__ wpack) {
    int tid = blockIdx.x * blockDim.x + threadIdx.x;
    if (tid >= WTOTAL) return;
    int isW2 = tid >= W1_ELEMS;
    int t0   = isW2 ? tid - W1_ELEMS : tid;
    int f    = t0 / FRAG_ELEMS;
    int r    = t0 % FRAG_ELEMS;
    int lane = r >> 4;
    int e    = r & 15;
    int nch  = isW2 ? 2 : 3;
    int t    = f / nch;
    int c    = f % nch;
    int n    = t * 16 + (lane & 15);
    int basek = 32 * c + ((lane >= 16) ? 8 : 0);
    int k    = basek + ((e < 8) ? e : (e + 8));
    float v;
    if (isW2) v = W2[k * HID + n];
    else      v = (k < DIN + 3) ? W1[k * HID + n] : 0.0f;  // pad K 67..95 with 0
    wpack[tid] = (__bf16)v;
}

__global__ void init_out_kernel(float* __restrict__ out, int n) {
    int i = blockIdx.x * blockDim.x + threadIdx.x;
    if (i < n) out[i] = -__builtin_inff();
}

// ---------------------------------------------------------------------------
// Main kernel: 8 waves / block, one 16-edge tile per wave.
// ---------------------------------------------------------------------------
__launch_bounds__(256)
__global__ void pointconv_kernel(const float* __restrict__ x,
                                 const float* __restrict__ pos,
                                 const int*   __restrict__ eidx,   // [2][E]
                                 const float* __restrict__ b1,
                                 const float* __restrict__ b2,
                                 const __bf16* __restrict__ wpack,
                                 float* __restrict__ out) {
    __shared__ __attribute__((aligned(32))) __bf16 ldsW[WTOTAL];        // 20KB
    __shared__ __attribute__((aligned(32))) __bf16 ldsH[8][16 * HID];   // 16KB

    int tid = threadIdx.x;
    // Cooperative weight stage: 20KB = 1280 x int4
    {
        const int4* s4 = (const int4*)wpack;
        int4* d4 = (int4*)ldsW;
        #pragma unroll
        for (int i = 0; i < (WTOTAL * 2 / 16) / 256; ++i)
            d4[tid + i * 256] = s4[tid + i * 256];
    }
    __syncthreads();

    int wave = tid >> 5;
    int lane = tid & 31;
    int tile = blockIdx.x * 8 + wave;
    if (tile >= NTILES) return;

    int l15 = lane & 15;
    int hi  = lane >> 4;                       // which K-half this lane holds
    int e   = tile * 16 + l15;                 // message id for this lane's row
    bool isE = e < E_EDGES;
    int es  = isE ? e : 0;
    int s   = isE ? eidx[es]            : (e - E_EDGES);
    int d   = isE ? eidx[E_EDGES + es]  : (e - E_EDGES);

    float rx = pos[s * 3 + 0] - pos[d * 3 + 0];
    float ry = pos[s * 3 + 1] - pos[d * 3 + 1];
    float rz = pos[s * 3 + 2] - pos[d * 3 + 2];

    // ---- Layer-1 A fragments (16x96 bf16, 3 chunks of K=32) ----
    v16bf a[3];
    const float* xr = x + s * DIN;
    #pragma unroll
    for (int c = 0; c < 2; ++c) {
        int g = 32 * c + hi * 8;
        float4 f0 = *(const float4*)(xr + g);
        float4 f1 = *(const float4*)(xr + g + 4);
        float4 f2 = *(const float4*)(xr + g + 16);
        float4 f3 = *(const float4*)(xr + g + 20);
        v16bf av;
        av[0]  = (__bf16)f0.x; av[1]  = (__bf16)f0.y; av[2]  = (__bf16)f0.z; av[3]  = (__bf16)f0.w;
        av[4]  = (__bf16)f1.x; av[5]  = (__bf16)f1.y; av[6]  = (__bf16)f1.z; av[7]  = (__bf16)f1.w;
        av[8]  = (__bf16)f2.x; av[9]  = (__bf16)f2.y; av[10] = (__bf16)f2.z; av[11] = (__bf16)f2.w;
        av[12] = (__bf16)f3.x; av[13] = (__bf16)f3.y; av[14] = (__bf16)f3.z; av[15] = (__bf16)f3.w;
        a[c] = av;
    }
    {   // chunk 2: K=64..95 -> rel(3) then zeros (branchless, EXEC stays full)
        v16bf av;
        #pragma unroll
        for (int i = 0; i < 16; ++i) av[i] = (__bf16)0.0f;
        av[0] = (__bf16)(hi ? 0.0f : rx);
        av[1] = (__bf16)(hi ? 0.0f : ry);
        av[2] = (__bf16)(hi ? 0.0f : rz);
        a[2] = av;
    }

    const v16bf* w1frag = (const v16bf*)ldsW;              // frag = 32 x v16bf
    const v16bf* w2frag = (const v16bf*)(ldsW + W1_ELEMS);
    __bf16* hrow = ldsH[wave];

    // ---- Layer 1: h1 = relu(A @ W1 + b1), staged to LDS row-major bf16 ----
    #pragma unroll
    for (int t = 0; t < 4; ++t) {
        float bv = b1[t * 16 + l15];
        v8f acc;
        #pragma unroll
        for (int r = 0; r < 8; ++r) acc[r] = bv;
        #pragma unroll
        for (int c = 0; c < 3; ++c) {
            v16bf bm = w1frag[(t * 3 + c) * 32 + lane];
            acc = __builtin_amdgcn_wmma_f32_16x16x32_bf16(
                false, a[c], false, bm, (short)0, acc, false, false);
        }
        // C layout: lane holds column N=t*16+l15, rows M = r + 8*hi
        #pragma unroll
        for (int r = 0; r < 8; ++r) {
            float v = fmaxf(acc[r], 0.0f);
            int M = r + hi * 8;
            hrow[M * HID + t * 16 + l15] = (__bf16)v;
        }
    }

    // ---- Transpose-read layer-2 A fragments from per-wave LDS staging ----
    v16bf h2[2];
    #pragma unroll
    for (int c = 0; c < 2; ++c) {
        int base = l15 * HID + 32 * c + hi * 8;
        v8bf lo = *(const v8bf*)(hrow + base);
        v8bf hh = *(const v8bf*)(hrow + base + 16);
        h2[c] = __builtin_shufflevector(lo, hh,
                                        0, 1, 2, 3, 4, 5, 6, 7,
                                        8, 9, 10, 11, 12, 13, 14, 15);
    }

    // ---- Layer 2 + segment-max scatter ----
    #pragma unroll
    for (int t = 0; t < 4; ++t) {
        float bv = b2[t * 16 + l15];
        v8f acc;
        #pragma unroll
        for (int r = 0; r < 8; ++r) acc[r] = bv;
        #pragma unroll
        for (int c = 0; c < 2; ++c) {
            v16bf bm = w2frag[(t * 2 + c) * 32 + lane];
            acc = __builtin_amdgcn_wmma_f32_16x16x32_bf16(
                false, h2[c], false, bm, (short)0, acc, false, false);
        }
        #pragma unroll
        for (int r = 0; r < 8; ++r) {
            int M  = r + hi * 8;                 // which edge row of the tile
            int dm = __shfl(d, M, 32);           // dst node of that edge
            atomicMax(out + dm * HID + t * 16 + l15, acc[r]);
        }
    }
}

// ---------------------------------------------------------------------------
extern "C" void kernel_launch(void* const* d_in, const int* in_sizes, int n_in,
                              void* d_out, int out_size, void* d_ws, size_t ws_size,
                              hipStream_t stream) {
    const float* x   = (const float*)d_in[0];
    const float* pos = (const float*)d_in[1];
    const int*   ei  = (const int*)  d_in[2];
    const float* W1  = (const float*)d_in[3];
    const float* b1  = (const float*)d_in[4];
    const float* W2  = (const float*)d_in[5];
    const float* b2  = (const float*)d_in[6];
    float* out = (float*)d_out;
    __bf16* wpack = (__bf16*)d_ws;

    pack_weights_kernel<<<(WTOTAL + 255) / 256, 256, 0, stream>>>(W1, W2, wpack);
    init_out_kernel<<<(N_NODES * HID + 255) / 256, 256, 0, stream>>>(out, N_NODES * HID);
    int blocks = (NTILES + 7) / 8;
    pointconv_kernel<<<blocks, 256, 0, stream>>>(x, pos, ei, b1, b2, wpack, out);
}
